// MBertSelfAttention_5239860101844
// MI455X (gfx1250) — matrix operand order repaired and checked by reading, not compiled, not run on hardware
//
#include <hip/hip_runtime.h>
#include <hip/hip_bf16.h>
#include <stdint.h>

// Problem constants (reference: B=4, T=2048, C=768, H=12, D=64)
#define Bn 4
#define Tn 2048
#define Cn 768
#define Hn 12
#define Dn 64
#define Mn (Bn*Tn)   // 8192 tokens

typedef __attribute__((ext_vector_type(16))) __bf16 v16bf;
typedef __attribute__((ext_vector_type(8)))  float  v8f;

union Frag { uint32_t u[8]; v16bf v; };

__device__ __forceinline__ uint16_t f32_to_bf16(float f) {
    uint32_t u = __float_as_uint(f);
    u += 0x7FFFu + ((u >> 16) & 1u);     // round-to-nearest-even
    return (uint16_t)(u >> 16);
}

__device__ __forceinline__ v8f v8f_zero() {
    v8f z;
#pragma unroll
    for (int i = 0; i < 8; i++) z[i] = 0.0f;
    return z;
}

__device__ __forceinline__ v8f wmma_bf16(const Frag& a, const Frag& b, v8f c) {
    // D = A(16x32 bf16) x B(32x16 bf16) + C(16x16 f32)
    return __builtin_amdgcn_wmma_f32_16x16x32_bf16(
        false, a.v, false, b.v, (short)0, c, false, false);
}

// CDNA5 async global->LDS copy (ASYNCcnt-tracked, bypasses VGPRs).
// GV addressing: vdst = LDS byte offset VGPR, vaddr = 64-bit global address.
__device__ __forceinline__ void async_load_b128(uint32_t lds_off, uint64_t gaddr) {
    asm volatile("global_load_async_to_lds_b128 %0, %1, off"
                 :: "v"(lds_off), "v"(gaddr) : "memory");
}
#define WAIT_ASYNC(n) asm volatile("s_wait_asynccnt " #n ::: "memory")

// ---------------------------------------------------------------------------
// fp32 -> bf16 conversion
// ---------------------------------------------------------------------------
__global__ void cvt_bf16_kernel(const float* __restrict__ in,
                                uint16_t* __restrict__ out, int n) {
    int i = blockIdx.x * blockDim.x + threadIdx.x;
    if (i < n) out[i] = f32_to_bf16(in[i]);
}

// ---------------------------------------------------------------------------
// QKV projection: y = x @ W^T + b  (M=8192, N=768, K=768), bf16 WMMA, f32 acc
// mode 0: Q (scaled by 1/sqrt(D), layout [bh,t,d])
// mode 1: K (layout [bh,t,d])
// mode 2: V (layout transposed [bh,d,t] for the PV matmul)
// Block = 128 threads (4 waves); block tile 64x64; wave tile 16x64.
// Double-buffered LDS fed by async-to-LDS loads.
// ---------------------------------------------------------------------------
__global__ __launch_bounds__(128) void qkv_gemm_kernel(
    const uint16_t* __restrict__ Xbf, const uint16_t* __restrict__ Wbf,
    const float* __restrict__ bias, uint16_t* __restrict__ out, int mode)
{
    __shared__ uint16_t lX[2][64][40];   // 64 rows x 32 k (+8 pad), 16B-aligned rows
    __shared__ uint16_t lW[2][64][40];

    const int tid  = threadIdx.x;
    const int wave = tid >> 5;
    const int lane = tid & 31;
    const int half = lane >> 4;
    const int lm   = lane & 15;
    const int m0   = blockIdx.x * 64;
    const int n0   = blockIdx.y * 64;

    // issue one stage (k-slab) of async copies: 4 x b128 per thread
    auto issue = [&](int buf, int ks) {
        const int k0 = ks * 32;
#pragma unroll
        for (int i = 0; i < 2; i++) {
            int c    = tid + 128 * i;   // 0..255 chunks of 8 elems
            int row  = c >> 2;          // 0..63
            int col8 = (c & 3) * 8;
            async_load_b128((uint32_t)(size_t)&lX[buf][row][col8],
                            (uint64_t)(size_t)(Xbf + (size_t)(m0 + row) * Cn + k0 + col8));
            async_load_b128((uint32_t)(size_t)&lW[buf][row][col8],
                            (uint64_t)(size_t)(Wbf + (size_t)(n0 + row) * Cn + k0 + col8));
        }
    };

    v8f acc[4];
#pragma unroll
    for (int jt = 0; jt < 4; jt++) acc[jt] = v8f_zero();

    issue(0, 0);
    for (int ks = 0; ks < Cn / 32; ks++) {
        const int cur = ks & 1;
        if (ks + 1 < Cn / 32) {
            issue(cur ^ 1, ks + 1);     // prefetch next slab into other buffer
            WAIT_ASYNC(0x4);            // current slab (older 4/thread) landed
        } else {
            WAIT_ASYNC(0x0);
        }
        __syncthreads();

        // A fragment: row = lane%16, K = half*16 + 2v pairs per VGPR (contiguous k order, same as B)
        Frag a;
        const int mrow = wave * 16 + lm;
#pragma unroll
        for (int v = 0; v < 8; v++) {
            int kk = (half << 4) + (v << 1);
            a.u[v] = *reinterpret_cast<const uint32_t*>(&lX[cur][mrow][kk]);
        }
#pragma unroll
        for (int jt = 0; jt < 4; jt++) {
            // B fragment: N = lane%16, K = half*16 + 2v (B element (k,n) = W[n][k])
            Frag fb;
#pragma unroll
            for (int v = 0; v < 8; v++) {
                int kk = (half << 4) + (v << 1);
                fb.u[v] = *reinterpret_cast<const uint32_t*>(&lW[cur][jt * 16 + lm][kk]);
            }
            acc[jt] = wmma_bf16(a, fb, acc[jt]);
        }
        __syncthreads();   // all waves done with buf[cur] before it is refilled
    }

    const float scale = (mode == 0) ? 0.125f : 1.0f;   // fold 1/sqrt(64) into Q
#pragma unroll
    for (int jt = 0; jt < 4; jt++) {
        int n_g = n0 + jt * 16 + lm;
        float bv = bias[n_g];
        int hh = n_g >> 6, d = n_g & 63;
#pragma unroll
        for (int r = 0; r < 8; r++) {
            int m_g = m0 + wave * 16 + r + half * 8;   // C/D layout: M = r + half*8
            int bb  = m_g >> 11, t = m_g & (Tn - 1);
            float val = (acc[jt][r] + bv) * scale;
            size_t idx;
            if (mode == 2) idx = (((size_t)bb * Hn + hh) * Dn + d) * Tn + t;  // V^T
            else           idx = (((size_t)bb * Hn + hh) * Tn + t) * Dn + d;  // Q/K
            out[idx] = f32_to_bf16(val);
        }
    }
}

// ---------------------------------------------------------------------------
// Fused flash attention: per block = 64 q rows of one (b,h); online softmax.
// Loops over 32 key tiles of 64; K tile and V^T tile double-buffered in LDS,
// filled by async-to-LDS copies overlapped with WMMA compute.
// ---------------------------------------------------------------------------
__global__ __launch_bounds__(128) void attn_kernel(
    const uint16_t* __restrict__ Qbf, const uint16_t* __restrict__ Kbf,
    const uint16_t* __restrict__ Vtbf, const float* __restrict__ mask,
    float* __restrict__ out)
{
    __shared__ uint16_t lK[2][64][72];     // 64 keys x 64 d   (+8 pad)
    __shared__ uint16_t lV[2][64][72];     // 64 d    x 64 keys(+8 pad)
    __shared__ uint16_t lP[4][16][72];     // per-wave P tile 16 x 64

    const int tid  = threadIdx.x;
    const int wave = tid >> 5;
    const int lane = tid & 31;
    const int half = lane >> 4;
    const int lm   = lane & 15;

    const int bh = blockIdx.y;
    const int b  = bh / Hn;
    const int h  = bh % Hn;
    const int q0 = blockIdx.x * 64;

    const uint16_t* Qh = Qbf  + (size_t)bh * Tn * Dn;
    const uint16_t* Kh = Kbf  + (size_t)bh * Tn * Dn;
    const uint16_t* Vh = Vtbf + (size_t)bh * Dn * Tn;
    const float*    mk = mask + (size_t)b * Tn;

    // issue one key-tile stage: 8 x b128 async copies per thread (K + V^T)
    auto issueKV = [&](int buf, int j) {
#pragma unroll
        for (int i = 0; i < 4; i++) {
            int c    = tid + 128 * i;   // 0..511 chunks of 8 elems
            int row  = c >> 3;          // 0..63
            int col8 = (c & 7) * 8;
            async_load_b128((uint32_t)(size_t)&lK[buf][row][col8],
                            (uint64_t)(size_t)(Kh + (size_t)(j * 64 + row) * Dn + col8));
            async_load_b128((uint32_t)(size_t)&lV[buf][row][col8],
                            (uint64_t)(size_t)(Vh + (size_t)row * Tn + j * 64 + col8));
        }
    };

    // Q A-fragments (K-dim = d, 64 = two 32-slabs), resident for whole kernel
    Frag aQ[2];
    {
        const int mrow = q0 + wave * 16 + lm;
#pragma unroll
        for (int s = 0; s < 2; s++)
#pragma unroll
            for (int v = 0; v < 8; v++) {
                int kk = s * 32 + (half << 4) + (v << 1);
                aQ[s].u[v] = *reinterpret_cast<const uint32_t*>(&Qh[(size_t)mrow * Dn + kk]);
            }
    }

    float mrun[8], lrun[8];
    v8f accO[4];
#pragma unroll
    for (int r = 0; r < 8; r++) { mrun[r] = -1e30f; lrun[r] = 0.0f; }
#pragma unroll
    for (int dt = 0; dt < 4; dt++) accO[dt] = v8f_zero();

    issueKV(0, 0);
    for (int j = 0; j < Tn / 64; j++) {
        const int cur = j & 1;
        if (j + 1 < Tn / 64) {
            issueKV(cur ^ 1, j + 1);    // prefetch next key tile
            WAIT_ASYNC(0x8);            // current tile (older 8/thread) landed
        } else {
            WAIT_ASYNC(0x0);
        }
        __syncthreads();

        // S = Q K^T : 16 q rows x 64 keys (4 n-tiles, K-dim d in 2 slabs)
        v8f S[4];
#pragma unroll
        for (int jt = 0; jt < 4; jt++) {
            v8f a = v8f_zero();
#pragma unroll
            for (int s = 0; s < 2; s++) {
                Frag fb;   // B element (k=d, n=key) = K[key][d]
#pragma unroll
                for (int v = 0; v < 8; v++)
                    fb.u[v] = *reinterpret_cast<const uint32_t*>(
                        &lK[cur][jt * 16 + lm][s * 32 + (half << 4) + (v << 1)]);
                a = wmma_bf16(aQ[s], fb, a);
            }
            S[jt] = a;
        }

        // additive mask bias: (1 - mask)*-1e4 per key column
#pragma unroll
        for (int jt = 0; jt < 4; jt++) {
            float bv = (1.0f - mk[j * 64 + jt * 16 + lm]) * -10000.0f;
#pragma unroll
            for (int r = 0; r < 8; r++) S[jt][r] += bv;
        }

        // online softmax: rows live across 16-lane halves (xor 1,2,4,8 stays in half)
#pragma unroll
        for (int r = 0; r < 8; r++) {
            float x = fmaxf(fmaxf(S[0][r], S[1][r]), fmaxf(S[2][r], S[3][r]));
            x = fmaxf(x, __shfl_xor(x, 1));
            x = fmaxf(x, __shfl_xor(x, 2));
            x = fmaxf(x, __shfl_xor(x, 4));
            x = fmaxf(x, __shfl_xor(x, 8));
            float mnew  = fmaxf(mrun[r], x);
            float alpha = __expf(mrun[r] - mnew);
            mrun[r] = mnew;
            lrun[r] *= alpha;
#pragma unroll
            for (int dt = 0; dt < 4; dt++) accO[dt][r] *= alpha;
        }

#pragma unroll
        for (int jt = 0; jt < 4; jt++)
#pragma unroll
            for (int r = 0; r < 8; r++)
                S[jt][r] = __expf(S[jt][r] - mrun[r]);

#pragma unroll
        for (int r = 0; r < 8; r++) {
            float s = S[0][r] + S[1][r] + S[2][r] + S[3][r];
            s += __shfl_xor(s, 1);
            s += __shfl_xor(s, 2);
            s += __shfl_xor(s, 4);
            s += __shfl_xor(s, 8);
            lrun[r] += s;
        }

        // P (C-layout) -> LDS bf16 -> re-read in A-layout (per-wave private,
        // same-wave DS ops are in order; compiler inserts s_wait_dscnt)
#pragma unroll
        for (int jt = 0; jt < 4; jt++)
#pragma unroll
            for (int r = 0; r < 8; r++)
                lP[wave][r + half * 8][jt * 16 + lm] = f32_to_bf16(S[jt][r]);

        // O += P @ V  (K-dim = key, 2 slabs; B element (k=key, n=d) = V^T[d][key])
#pragma unroll
        for (int s = 0; s < 2; s++) {
            Frag aP;
#pragma unroll
            for (int v = 0; v < 8; v++) {
                int kk = s * 32 + (half << 4) + (v << 1);
                aP.u[v] = *reinterpret_cast<const uint32_t*>(&lP[wave][lm][kk]);
            }
#pragma unroll
            for (int dt = 0; dt < 4; dt++) {
                Frag fb;
#pragma unroll
                for (int v = 0; v < 8; v++)
                    fb.u[v] = *reinterpret_cast<const uint32_t*>(
                        &lV[cur][dt * 16 + lm][s * 32 + (half << 4) + (v << 1)]);
                accO[dt] = wmma_bf16(aP, fb, accO[dt]);
            }
        }
        __syncthreads();   // all waves done with buf[cur] before it is refilled
    }

    // epilogue: ctx [B,T,C] fp32
#pragma unroll
    for (int dt = 0; dt < 4; dt++) {
#pragma unroll
        for (int r = 0; r < 8; r++) {
            int m_g = q0 + wave * 16 + r + half * 8;
            int cg  = h * Dn + dt * 16 + lm;
            out[((size_t)b * Tn + m_g) * Cn + cg] = accO[dt][r] / lrun[r];
        }
    }
}

// ---------------------------------------------------------------------------
extern "C" void kernel_launch(void* const* d_in, const int* in_sizes, int n_in,
                              void* d_out, int out_size, void* d_ws, size_t ws_size,
                              hipStream_t stream)
{
    (void)in_sizes; (void)n_in; (void)out_size; (void)ws_size;
    const float* X    = (const float*)d_in[0];
    const float* mask = (const float*)d_in[1];
    const float* Wq   = (const float*)d_in[2];
    const float* bq   = (const float*)d_in[3];
    const float* Wk   = (const float*)d_in[4];
    const float* bk   = (const float*)d_in[5];
    const float* Wv   = (const float*)d_in[6];
    const float* bv   = (const float*)d_in[7];
    float* out = (float*)d_out;

    // Workspace layout (~54 MB total)
    uint8_t* ws = (uint8_t*)d_ws;
    size_t off = 0;
    uint16_t* Xbf = (uint16_t*)(ws + off); off += (size_t)Mn * Cn * 2;
    uint16_t* Wqb = (uint16_t*)(ws + off); off += (size_t)Cn * Cn * 2;
    uint16_t* Wkb = (uint16_t*)(ws + off); off += (size_t)Cn * Cn * 2;
    uint16_t* Wvb = (uint16_t*)(ws + off); off += (size_t)Cn * Cn * 2;
    uint16_t* Qbf = (uint16_t*)(ws + off); off += (size_t)Mn * Cn * 2;
    uint16_t* Kbf = (uint16_t*)(ws + off); off += (size_t)Mn * Cn * 2;
    uint16_t* Vtb = (uint16_t*)(ws + off); off += (size_t)Mn * Cn * 2;

    {
        int n = Mn * Cn;
        cvt_bf16_kernel<<<(n + 255) / 256, 256, 0, stream>>>(X, Xbf, n);
        int nw = Cn * Cn;
        cvt_bf16_kernel<<<(nw + 255) / 256, 256, 0, stream>>>(Wq, Wqb, nw);
        cvt_bf16_kernel<<<(nw + 255) / 256, 256, 0, stream>>>(Wk, Wkb, nw);
        cvt_bf16_kernel<<<(nw + 255) / 256, 256, 0, stream>>>(Wv, Wvb, nw);
    }

    dim3 gq(Mn / 64, Cn / 64);   // 128 x 12
    qkv_gemm_kernel<<<gq, 128, 0, stream>>>(Xbf, Wqb, bq, Qbf, 0);
    qkv_gemm_kernel<<<gq, 128, 0, stream>>>(Xbf, Wkb, bk, Kbf, 1);
    qkv_gemm_kernel<<<gq, 128, 0, stream>>>(Xbf, Wvb, bv, Vtb, 2);

    dim3 ga(Tn / 64, Bn * Hn);   // 32 x 48
    attn_kernel<<<ga, 128, 0, stream>>>(Qbf, Kbf, Vtb, mask, out);
}
